// Interaction_Block_82429012345240
// MI455X (gfx1250) — matrix operand order repaired and checked by reading, not compiled, not run on hardware
//
#include <hip/hip_runtime.h>

typedef _Float16 h16;
typedef __attribute__((ext_vector_type(16))) _Float16 v16h;
typedef __attribute__((ext_vector_type(8)))  float v8f;
typedef __attribute__((ext_vector_type(4)))  float f4;

#define NN 20000
#define NE 640000
#define LDIM 128
#define TILE_E 64   // edges per workgroup (4 row-tiles of 16)
#define TILE_N 32   // nodes per workgroup (2 row-tiles of 16)

// ---------------- input indices (setup_inputs order; params flattened with
// sorted dict keys at each level, leaves b1,b2[,bln,g],w1,w2) ----------------
enum {
  IN_EDGE_INDEX = 0, IN_SPOS, IN_RPOS, IN_DX, IN_EATTR, IN_VA, IN_VB, IN_VC,
  IN_SVT, IN_SWT, IN_RVT, IN_RWT, IN_NLAT, IN_VEL,
  P_CC_B1 = 14, P_CC_B2, P_CC_W1, P_CC_W2,
  P_EE_B1, P_EE_B2, P_EE_BLN, P_EE_G, P_EE_W1, P_EE_W2,
  P_EI_B1, P_EI_B2, P_EI_BLN, P_EI_G, P_EI_W1, P_EI_W2,
  P_EN_B1, P_EN_B2, P_EN_BLN, P_EN_G, P_EN_W1, P_EN_W2,
  P_ED_B1, P_ED_B2, P_ED_W1, P_ED_W2,
  P_FS_B1, P_FS_B2, P_FS_W1, P_FS_W2,
  P_I1_B1, P_I1_B2, P_I1_W1, P_I1_W2,
  P_I2_B1, P_I2_B2, P_I2_W1, P_I2_W2,
  P_II_B1, P_II_B2, P_II_W1, P_II_W2,
  P_IM_B1, P_IM_B2, P_IM_W1, P_IM_W2,
  P_NW_B1, P_NW_B2, P_NW_W1, P_NW_W2,
  P_VS_B1, P_VS_B2, P_VS_W1, P_VS_W2,
};

// ---------------- workspace layout ----------------
constexpr size_t OFF_FORCE  = 0;                       // N*3 (zeroed)
constexpr size_t OFF_TORQUE = OFF_FORCE  + 3 * NN;     // N*3 (zeroed)
constexpr size_t OFF_STATIC = OFF_TORQUE + 3 * NN;     // N*3 (zeroed)
constexpr size_t OFF_COUNT  = OFF_STATIC + 3 * NN;     // N   (zeroed)
constexpr size_t OFF_WNODE  = OFF_COUNT  + NN;         // N
constexpr size_t OFF_INVM   = OFF_WNODE  + NN;         // N
constexpr size_t OFF_INVI   = OFF_INVM   + NN;         // N
constexpr size_t OFF_EXTDV  = OFF_INVI   + NN;         // N*3
constexpr size_t OFF_VSC    = OFF_EXTDV  + 3 * NN;     // N
constexpr size_t F32_TOTAL  = OFF_VSC    + NN;
constexpr size_t F16_BASE_BYTES = ((F32_TOTAL * 4 + 255) / 256) * 256;

// f16 weight fragment buffers (units: halves). frag = Ktiles*Ntiles*512 halves
constexpr size_t HW_ENCN1 = 0;
constexpr size_t HW_ENCN2 = HW_ENCN1 + 1 * 8 * 512;
constexpr size_t HW_ENCE1 = HW_ENCN2 + 4 * 8 * 512;
constexpr size_t HW_ENCE2 = HW_ENCE1 + 1 * 8 * 512;
constexpr size_t HW_ENCI1 = HW_ENCE2 + 4 * 8 * 512;
constexpr size_t HW_ENCI2 = HW_ENCI1 + 12 * 8 * 512;
constexpr size_t HW_I1W1  = HW_ENCI2 + 4 * 8 * 512;
constexpr size_t HW_I1W2  = HW_I1W1  + 4 * 8 * 512;
constexpr size_t HW_I2W1  = HW_I1W2  + 4 * 1 * 512;
constexpr size_t HW_I2W2  = HW_I2W1  + 4 * 8 * 512;
constexpr size_t HW_FSW1  = HW_I2W2  + 4 * 1 * 512;
constexpr size_t HW_FSW2  = HW_FSW1  + 4 * 8 * 512;
constexpr size_t HW_CCW1  = HW_FSW2  + 4 * 1 * 512;
constexpr size_t HW_CCW2  = HW_CCW1  + 4 * 8 * 512;
constexpr size_t HW_NWW1  = HW_CCW2  + 4 * 1 * 512;
constexpr size_t HW_NWW2  = HW_NWW1  + 4 * 8 * 512;
constexpr size_t HW_IMW1  = HW_NWW2  + 4 * 1 * 512;
constexpr size_t HW_IMW2  = HW_IMW1  + 4 * 8 * 512;
constexpr size_t HW_IIW1  = HW_IMW2  + 4 * 1 * 512;
constexpr size_t HW_IIW2  = HW_IIW1  + 4 * 8 * 512;
constexpr size_t HW_EDW1  = HW_IIW2  + 4 * 1 * 512;
constexpr size_t HW_EDW2  = HW_EDW1  + 4 * 8 * 512;
constexpr size_t HW_VSW1  = HW_EDW2  + 4 * 1 * 512;
constexpr size_t HW_VSW2  = HW_VSW1  + 4 * 8 * 512;

// ---------------- WMMA fragment helpers ----------------
// B fragment (32x16 K-tile): N=lane%16; K = 8*(lane/16) + (e<8 ? e : e+8).
// Pre-swizzled: 512 halves/tile, 16 contiguous halves per lane.
__device__ inline v16h load_bfrag(const h16* base, int tile) {
  int lane = threadIdx.x & 31;
  union { v16h v; f4 q[2]; } u;
  const f4* p = (const f4*)(base + ((size_t)tile << 9) + (lane << 4));
  u.q[0] = p[0];
  u.q[1] = p[1];
  return u.v;
}

// A fragment (16x32) from row-major f16 LDS tile: lane%16 = row, lanes>=16
// offset K by 8; element e -> K = off + (e<8 ? e : e+8): two 8-half chunks.
__device__ inline v16h lds_afrag(const h16* x, int stride, int kbase) {
  int lane = threadIdx.x & 31;
  int row  = lane & 15;
  int off  = (lane & 16) ? 8 : 0;
  const h16* p = x + row * stride + kbase + off;
  union { v16h v; f4 q[2]; } u;
  u.q[0] = *(const f4*)(p);
  u.q[1] = *(const f4*)(p + 16);
  return u.v;
}

// A fragment from a tiny (K<=8) f32 feature tile [16][8], K padded to 32.
__device__ inline v16h afrag_small(const float* feat8) {
  int lane = threadIdx.x & 31;
  int row  = lane & 15;
  int off  = (lane & 16) ? 8 : 0;
  v16h a;
#pragma unroll
  for (int e = 0; e < 16; e++) {
    int K = off + (e < 8 ? e : e + 8);
    a[e] = (K < 8) ? (h16)feat8[row * 8 + K] : (h16)0.f;
  }
  return a;
}

__device__ inline v8f wmma16(v16h a, v16h b, v8f c) {
  return __builtin_amdgcn_wmma_f32_16x16x32_f16(false, a, false, b, (short)0, c,
                                                false, false);
}

// C/D tile layout: vgpr r -> row r + 8*(lane/16), col = lane%16 (+ 16*nt)
__device__ inline void store_h_relu(h16* hdst, int nt, v8f acc, const float* b1) {
  int lane = threadIdx.x & 31;
  int col = nt * 16 + (lane & 15);
  int rbase = (lane & 16) ? 8 : 0;
  float bias = b1[col];
#pragma unroll
  for (int r = 0; r < 8; r++) {
    float v = acc[r] + bias;
    hdst[(rbase + r) * LDIM + col] = (h16)fmaxf(v, 0.f);
  }
}

__device__ inline void store_y(float* y, int nt, v8f acc, const float* b2) {
  int lane = threadIdx.x & 31;
  int col = nt * 16 + (lane & 15);
  int rbase = (lane & 16) ? 8 : 0;
  float bias = b2[col];
#pragma unroll
  for (int r = 0; r < 8; r++) y[(rbase + r) * LDIM + col] = acc[r] + bias;
}

// Cooperative LayerNorm over a 64x128 f32 LDS tile; writes f16 (optionally
// accumulating) into dst, optionally f32 rows to global gout.
__device__ void layernorm64(const float* y, float* mu, float* rstd,
                            const float* g, const float* bln, h16* dst,
                            int dstride, int dcol0, bool accum, float* gout) {
  int tid = threadIdx.x;
  __syncthreads();  // y complete
  if (tid < TILE_E) {
    float s = 0.f, s2 = 0.f;
#pragma unroll 8
    for (int c = 0; c < LDIM; c++) {
      float v = y[tid * LDIM + c];
      s += v;
      s2 += v * v;
    }
    float m = s * (1.f / LDIM);
    mu[tid] = m;
    rstd[tid] = rsqrtf(fmaxf(s2 * (1.f / LDIM) - m * m, 0.f) + 1e-5f);
  }
  __syncthreads();
  int row = tid >> 2, c0 = (tid & 3) << 5;  // 4 threads/row, 32 cols each
  float m = mu[row], rs = rstd[row];
#pragma unroll 8
  for (int i = 0; i < 32; i++) {
    int c = c0 + i;
    float v = (y[row * LDIM + c] - m) * rs * g[c] + bln[c];
    h16* dp = dst + row * dstride + dcol0 + c;
    if (accum) *dp = (h16)((float)*dp + v);
    else       *dp = (h16)v;
    if (gout) gout[row * LDIM + c] = v;
  }
  __syncthreads();  // allow reuse of y / dst buffers
}

// ---------------- kernel 0: weight fp32 -> f16 fragment swizzle ----------------
__global__ __launch_bounds__(256) void convw_kernel(const float* src, h16* dst,
                                                    int Kact, int Ktiles,
                                                    int Ntiles, int Nact) {
  long total = (long)Ktiles * Ntiles * 512;
  for (long idx = (long)blockIdx.x * blockDim.x + threadIdx.x; idx < total;
       idx += (long)gridDim.x * blockDim.x) {
    long tile = idx >> 9;
    int r = (int)(idx & 511);
    int lane = r >> 4, e = r & 15;
    int kt = (int)(tile / Ntiles), nt = (int)(tile % Ntiles);
    int K = kt * 32 + ((lane & 16) ? 8 : 0) + (e < 8 ? e : e + 8);
    int col = nt * 16 + (lane & 15);
    float v = (K < Kact && col < Nact) ? src[(size_t)K * Nact + col] : 0.f;
    dst[idx] = (h16)v;
  }
}

__global__ __launch_bounds__(256) void zero_kernel(float* p, int n) {
  for (int i = blockIdx.x * blockDim.x + threadIdx.x; i < n;
       i += gridDim.x * blockDim.x)
    p[i] = 0.f;
}

// ---------------- kernel 1: per-node MLPs (32 nodes/WG, 2 row-tiles) ----------
struct NodeArgs {
  const float* nlat;
  const h16* w1[5];
  const h16* w2[5];
  const float* b1[5];
  const float* b2[5];
  float* outp[5];  // wnode, inv_mass, inv_inertia, external_dv(3), vel_scaler
};

__global__ __launch_bounds__(256) void node_kernel(NodeArgs a) {
  __shared__ h16 xl[TILE_N * LDIM];
  __shared__ h16 hl[TILE_N * LDIM];
  int tid = threadIdx.x, wave = tid >> 5, lane = tid & 31;
  int n0 = blockIdx.x * TILE_N;
  {
    int row = tid >> 3, c0 = (tid & 7) << 4;  // 8 threads/row, 16 cols each
    const float* p = a.nlat + (size_t)(n0 + row) * LDIM + c0;
#pragma unroll
    for (int i = 0; i < 16; i++) xl[row * LDIM + c0 + i] = (h16)p[i];
  }
  __syncthreads();
#pragma unroll 1
  for (int m = 0; m < 5; m++) {
    v8f acc[2] = {};
#pragma unroll
    for (int kt = 0; kt < 4; kt++) {
      v16h b = load_bfrag(a.w1[m], kt * 8 + wave);
#pragma unroll
      for (int rt = 0; rt < 2; rt++)
        acc[rt] = wmma16(lds_afrag(xl + rt * 16 * LDIM, LDIM, kt * 32), b, acc[rt]);
    }
#pragma unroll
    for (int rt = 0; rt < 2; rt++)
      store_h_relu(hl + rt * 16 * LDIM, wave, acc[rt], a.b1[m]);
    __syncthreads();
    if (wave < 2) {  // row-tile rt = wave
      v8f a2 = {};
#pragma unroll
      for (int kt = 0; kt < 4; kt++)
        a2 = wmma16(lds_afrag(hl + wave * 16 * LDIM, LDIM, kt * 32),
                    load_bfrag(a.w2[m], kt), a2);
      int dout = (m == 3) ? 3 : 1;
      int col = lane & 15, rbase = (lane & 16) ? 8 : 0;
      if (col < dout) {
        float bias = a.b2[m][col];
        float* out = a.outp[m];
#pragma unroll
        for (int r = 0; r < 8; r++) {
          int n = n0 + wave * 16 + rbase + r;
          out[(size_t)n * dout + col] = a2[r] + bias;
        }
      }
    }
    __syncthreads();
  }
}

// ---------------- kernel 2: per-edge pipeline (64 edges/WG) ----------------
struct EdgeArgs {
  const int *senders, *receivers;
  const float *spos, *rpos, *dx, *eattr, *va, *vb, *vc, *svt, *swt, *rvt, *rwt, *nlat;
  const float *en_b1, *en_b2, *en_g, *en_bln;
  const float *ee_b1, *ee_b2, *ee_g, *ee_bln;
  const float *ei_b1, *ei_b2, *ei_g, *ei_bln;
  const float *i1_b1, *i1_b2, *i2_b1, *i2_b2, *fs_b1, *fs_b2, *cc_b1, *cc_b2;
  const h16 *wENCN1, *wENCN2, *wENCE1, *wENCE2, *wENCI1, *wENCI2;
  const h16 *wI1a, *wI1b, *wI2a, *wI2b, *wFSa, *wFSb, *wCCa, *wCCb;
  const float* wnode;
  float *accF, *accT, *accS, *accC;
  float* out_int;  // interaction_latent [E,128] f32
};

__device__ void encoder_pass(const float* feat8, const h16* w1f, const h16* w2f,
                             const float* b1, const float* b2, const float* g,
                             const float* bln, h16* hbuf, float* ylds,
                             float* mu, float* rstd, h16* dst, int dstride,
                             int dcol0, bool accum) {
  int wave = threadIdx.x >> 5;
  {
    v16h b = load_bfrag(w1f, wave);  // K padded to 32, single K-tile
#pragma unroll
    for (int rt = 0; rt < 4; rt++) {
      v8f acc = {};
      acc = wmma16(afrag_small(feat8 + rt * 16 * 8), b, acc);
      store_h_relu(hbuf + rt * 16 * LDIM, wave, acc, b1);
    }
  }
  __syncthreads();
  {
    v8f acc[4] = {};
#pragma unroll
    for (int kt = 0; kt < 4; kt++) {
      v16h b = load_bfrag(w2f, kt * 8 + wave);
#pragma unroll
      for (int rt = 0; rt < 4; rt++)
        acc[rt] = wmma16(lds_afrag(hbuf + rt * 16 * LDIM, LDIM, kt * 32), b, acc[rt]);
    }
#pragma unroll
    for (int rt = 0; rt < 4; rt++)
      store_y(ylds + rt * 16 * LDIM, wave, acc[rt], b2);
  }
  layernorm64(ylds, mu, rstd, g, bln, dst, dstride, dcol0, accum, nullptr);
}

__global__ __launch_bounds__(256) void edge_kernel(EdgeArgs a) {
  __shared__ float sfeat[TILE_E * 8], rfeat[TILE_E * 8], efeat[TILE_E * 8];
  __shared__ int sid[TILE_E], rid[TILE_E];
  __shared__ h16 int_in[TILE_E * 384];   // [s+r lat | nl(s)+nl(r) | edge lat]
  __shared__ h16 intlat[TILE_E * LDIM];
  __shared__ h16 hbuf[TILE_E * LDIM];
  __shared__ float ylds[TILE_E * LDIM];
  __shared__ float mu_s[TILE_E], rstd_s[TILE_E];
  __shared__ float coef[4][TILE_E * 4];

  int tid = threadIdx.x, wave = tid >> 5, lane = tid & 31;
  size_t e0 = (size_t)blockIdx.x * TILE_E;

  // --- phase 1: scalar per-edge features (one edge per lane) ---
  if (tid < TILE_E) {
    size_t e = e0 + tid;
    float A0 = a.va[e * 3 + 0], A1 = a.va[e * 3 + 1], A2 = a.va[e * 3 + 2];
    float B0 = a.vb[e * 3 + 0], B1 = a.vb[e * 3 + 1], B2 = a.vb[e * 3 + 2];
    float C0 = a.vc[e * 3 + 0], C1 = a.vc[e * 3 + 1], C2 = a.vc[e * 3 + 2];
    float sv0 = a.svt[e * 3 + 0], sv1 = a.svt[e * 3 + 1], sv2 = a.svt[e * 3 + 2];
    float sw0 = a.swt[e * 3 + 0], sw1 = a.swt[e * 3 + 1], sw2 = a.swt[e * 3 + 2];
    float rv0 = a.rvt[e * 3 + 0], rv1 = a.rvt[e * 3 + 1], rv2 = a.rvt[e * 3 + 2];
    float rw0 = a.rwt[e * 3 + 0], rw1 = a.rwt[e * 3 + 1], rw2 = a.rwt[e * 3 + 2];
    sfeat[tid * 8 + 0] = A0 * sv0 + A1 * sv1 + A2 * sv2;
    sfeat[tid * 8 + 1] = B0 * sv0 + B1 * sv1 + B2 * sv2;
    sfeat[tid * 8 + 2] = C0 * sv0 + C1 * sv1 + C2 * sv2;
    sfeat[tid * 8 + 3] = A0 * sw0 + A1 * sw1 + A2 * sw2;
    sfeat[tid * 8 + 4] = B0 * sw0 + B1 * sw1 + B2 * sw2;
    sfeat[tid * 8 + 5] = C0 * sw0 + C1 * sw1 + C2 * sw2;
    sfeat[tid * 8 + 6] = 0.f; sfeat[tid * 8 + 7] = 0.f;
    rfeat[tid * 8 + 0] = -(A0 * rv0 + A1 * rv1 + A2 * rv2);
    rfeat[tid * 8 + 1] = -(B0 * rv0 + B1 * rv1 + B2 * rv2);
    rfeat[tid * 8 + 2] = -(C0 * rv0 + C1 * rv1 + C2 * rv2);
    rfeat[tid * 8 + 3] = -(A0 * rw0 + A1 * rw1 + A2 * rw2);
    rfeat[tid * 8 + 4] = -(B0 * rw0 + B1 * rw1 + B2 * rw2);
    rfeat[tid * 8 + 5] = -(C0 * rw0 + C1 * rw1 + C2 * rw2);
    rfeat[tid * 8 + 6] = 0.f; rfeat[tid * 8 + 7] = 0.f;
    float d0 = a.dx[e * 3 + 0], d1 = a.dx[e * 3 + 1], d2 = a.dx[e * 3 + 2];
    efeat[tid * 8 + 0] = sqrtf(d0 * d0 + d1 * d1 + d2 * d2);
    efeat[tid * 8 + 1] = a.eattr[e * 3 + 0];
    efeat[tid * 8 + 2] = a.eattr[e * 3 + 1];
    efeat[tid * 8 + 3] = a.eattr[e * 3 + 2];
    efeat[tid * 8 + 4] = 0.f; efeat[tid * 8 + 5] = 0.f;
    efeat[tid * 8 + 6] = 0.f; efeat[tid * 8 + 7] = 0.f;
    sid[tid] = a.senders[e];
    rid[tid] = a.receivers[e];
  }
  __syncthreads();

  // gather node_latent[s]+node_latent[r] into int_in cols [128,256)
  {
    int row = tid >> 2, c0 = (tid & 3) << 5;  // 4 threads/row, 32 cols
    const float* ps = a.nlat + (size_t)sid[row] * LDIM + c0;
    const float* pr = a.nlat + (size_t)rid[row] * LDIM + c0;
#pragma unroll 8
    for (int i = 0; i < 32; i++)
      int_in[row * 384 + 128 + c0 + i] = (h16)(ps[i] + pr[i]);
  }

  // --- phase 2: encoders (with LayerNorm) ---
  encoder_pass(sfeat, a.wENCN1, a.wENCN2, a.en_b1, a.en_b2, a.en_g, a.en_bln,
               hbuf, ylds, mu_s, rstd_s, int_in, 384, 0, false);
  encoder_pass(rfeat, a.wENCN1, a.wENCN2, a.en_b1, a.en_b2, a.en_g, a.en_bln,
               hbuf, ylds, mu_s, rstd_s, int_in, 384, 0, true);
  encoder_pass(efeat, a.wENCE1, a.wENCE2, a.ee_b1, a.ee_b2, a.ee_g, a.ee_bln,
               hbuf, ylds, mu_s, rstd_s, int_in, 384, 256, false);

  // --- phase 3: interaction encoder (K=384) ---
  {
    v8f acc[4] = {};
#pragma unroll
    for (int kt = 0; kt < 12; kt++) {
      v16h b = load_bfrag(a.wENCI1, kt * 8 + wave);
#pragma unroll
      for (int rt = 0; rt < 4; rt++)
        acc[rt] = wmma16(lds_afrag(int_in + rt * 16 * 384, 384, kt * 32), b, acc[rt]);
    }
#pragma unroll
    for (int rt = 0; rt < 4; rt++)
      store_h_relu(hbuf + rt * 16 * LDIM, wave, acc[rt], a.ei_b1);
    __syncthreads();
    v8f acc2[4] = {};
#pragma unroll
    for (int kt = 0; kt < 4; kt++) {
      v16h b = load_bfrag(a.wENCI2, kt * 8 + wave);
#pragma unroll
      for (int rt = 0; rt < 4; rt++)
        acc2[rt] = wmma16(lds_afrag(hbuf + rt * 16 * LDIM, LDIM, kt * 32), b, acc2[rt]);
    }
#pragma unroll
    for (int rt = 0; rt < 4; rt++)
      store_y(ylds + rt * 16 * LDIM, wave, acc2[rt], a.ei_b2);
    layernorm64(ylds, mu_s, rstd_s, a.ei_g, a.ei_bln, intlat, LDIM, 0, false,
                a.out_int + e0 * LDIM);
  }

  // --- phase 4: decoders i1, i2, f_scaler, comp_corr ---
  {
    const h16* W1s[4] = {a.wI1a, a.wI2a, a.wFSa, a.wCCa};
    const h16* W2s[4] = {a.wI1b, a.wI2b, a.wFSb, a.wCCb};
    const float* B1s[4] = {a.i1_b1, a.i2_b1, a.fs_b1, a.cc_b1};
    const float* B2s[4] = {a.i1_b2, a.i2_b2, a.fs_b2, a.cc_b2};
    const int douts[4] = {3, 3, 1, 3};
#pragma unroll
    for (int d = 0; d < 4; d++) {
      v8f acc[4] = {};
#pragma unroll
      for (int kt = 0; kt < 4; kt++) {
        v16h b = load_bfrag(W1s[d], kt * 8 + wave);
#pragma unroll
        for (int rt = 0; rt < 4; rt++)
          acc[rt] = wmma16(lds_afrag(intlat + rt * 16 * LDIM, LDIM, kt * 32), b, acc[rt]);
      }
#pragma unroll
      for (int rt = 0; rt < 4; rt++)
        store_h_relu(hbuf + rt * 16 * LDIM, wave, acc[rt], B1s[d]);
      __syncthreads();
      if (wave < 4) {  // row-tile rt = wave, single 16-col output tile
        v8f a2 = {};
#pragma unroll
        for (int kt = 0; kt < 4; kt++)
          a2 = wmma16(lds_afrag(hbuf + wave * 16 * LDIM, LDIM, kt * 32),
                      load_bfrag(W2s[d], kt), a2);
        int col = lane & 15, rbase = (lane & 16) ? 8 : 0;
        if (col < 4) {
          float bias = (col < douts[d]) ? B2s[d][col] : 0.f;
#pragma unroll
          for (int r = 0; r < 8; r++)
            coef[d][(wave * 16 + rbase + r) * 4 + col] = a2[r] + bias;
        }
      }
      __syncthreads();
    }
  }

  // --- phase 5: epilogue + scatter (segment sums via atomics) ---
  if (tid < TILE_E) {
    size_t e = e0 + tid;
    float A[3], B[3], C[3];
#pragma unroll
    for (int d = 0; d < 3; d++) {
      A[d] = a.va[e * 3 + d];
      B[d] = a.vb[e * 3 + d];
      C[d] = a.vc[e * 3 + d];
    }
    float cf0 = coef[0][tid * 4 + 0], cf1 = coef[0][tid * 4 + 1], cf2 = coef[0][tid * 4 + 2];
    float ca0 = coef[1][tid * 4 + 0], ca1 = coef[1][tid * 4 + 1], ca2 = coef[1][tid * 4 + 2];
    float lam = coef[2][tid * 4 + 0];
    float cc0 = coef[3][tid * 4 + 0], cc1 = coef[3][tid * 4 + 1], cc2 = coef[3][tid * 4 + 2];
    float f[3], aij[3], dxij[3];
#pragma unroll
    for (int d = 0; d < 3; d++) {
      f[d]    = cf0 * A[d] + cf1 * B[d] + cf2 * C[d];
      aij[d]  = ca0 * A[d] + ca1 * B[d] + ca2 * C[d];
      dxij[d] = cc0 * A[d] + cc1 * B[d] + cc2 * C[d];
    }
    int s = sid[tid], r = rid[tid];
    float ws = a.wnode[s], wr = a.wnode[r];
    float inv = 1.f / (ws + wr);
    float lever[3], fl[3];
#pragma unroll
    for (int d = 0; d < 3; d++) {
      float sp = a.spos[e * 3 + d], rp = a.rpos[e * 3 + d];
      float r0 = (ws * sp + wr * rp) * inv;
      lever[d] = rp - r0;
      fl[d] = f[d] * lam;
    }
    float tau[3];
    tau[0] = aij[0] - (lever[1] * fl[2] - lever[2] * fl[1]);
    tau[1] = aij[1] - (lever[2] * fl[0] - lever[0] * fl[2]);
    tau[2] = aij[2] - (lever[0] * fl[1] - lever[1] * fl[0]);
#pragma unroll
    for (int d = 0; d < 3; d++) {
      atomicAdd(&a.accF[(size_t)r * 3 + d], f[d]);
      atomicAdd(&a.accT[(size_t)r * 3 + d], tau[d]);
      atomicAdd(&a.accS[(size_t)r * 3 + d], dxij[d]);
    }
    atomicAdd(&a.accC[r], 1.f);
  }
}

// ---------------- kernel 3: finalize per-node outputs ----------------
struct FinArgs {
  const float *force, *torque, *stat, *count, *invm, *invi, *extdv, *vsc, *vel;
  float* out;
};

__global__ __launch_bounds__(256) void finalize_kernel(FinArgs a) {
  int i = blockIdx.x * blockDim.x + threadIdx.x;
  if (i >= NN) return;
  float cnt = fmaxf(a.count[i], 1.f);
  float im = a.invm[i], ii = a.invi[i], vs = a.vsc[i];
#pragma unroll
  for (int d = 0; d < 3; d++) {
    a.out[(size_t)i * 3 + d] = im * a.force[(size_t)i * 3 + d];
    a.out[(size_t)3 * NN + i * 3 + d] = ii * a.torque[(size_t)i * 3 + d];
    a.out[(size_t)6 * NN + i * 3 + d] =
        (a.vel[(size_t)i * 3 + d] + a.extdv[(size_t)i * 3 + d]) * vs +
        a.stat[(size_t)i * 3 + d] / cnt;
  }
}

// ---------------- host launch ----------------
struct ConvDesc { int src; size_t dst; int Kact, Ktiles, Nact; };

extern "C" void kernel_launch(void* const* d_in, const int* in_sizes, int n_in,
                              void* d_out, int out_size, void* d_ws,
                              size_t ws_size, hipStream_t stream) {
  (void)in_sizes; (void)n_in; (void)out_size; (void)ws_size;
  char* ws = (char*)d_ws;
  float* fws = (float*)ws;
  h16* hws = (h16*)(ws + F16_BASE_BYTES);
#define FIN(i) ((const float*)d_in[(i)])

  // weight conversion + fragment swizzle
  const ConvDesc convs[] = {
      {P_EN_W1, HW_ENCN1, 6, 1, 128},   {P_EN_W2, HW_ENCN2, 128, 4, 128},
      {P_EE_W1, HW_ENCE1, 4, 1, 128},   {P_EE_W2, HW_ENCE2, 128, 4, 128},
      {P_EI_W1, HW_ENCI1, 384, 12, 128},{P_EI_W2, HW_ENCI2, 128, 4, 128},
      {P_I1_W1, HW_I1W1, 128, 4, 128},  {P_I1_W2, HW_I1W2, 128, 4, 3},
      {P_I2_W1, HW_I2W1, 128, 4, 128},  {P_I2_W2, HW_I2W2, 128, 4, 3},
      {P_FS_W1, HW_FSW1, 128, 4, 128},  {P_FS_W2, HW_FSW2, 128, 4, 1},
      {P_CC_W1, HW_CCW1, 128, 4, 128},  {P_CC_W2, HW_CCW2, 128, 4, 3},
      {P_NW_W1, HW_NWW1, 128, 4, 128},  {P_NW_W2, HW_NWW2, 128, 4, 1},
      {P_IM_W1, HW_IMW1, 128, 4, 128},  {P_IM_W2, HW_IMW2, 128, 4, 1},
      {P_II_W1, HW_IIW1, 128, 4, 128},  {P_II_W2, HW_IIW2, 128, 4, 1},
      {P_ED_W1, HW_EDW1, 128, 4, 128},  {P_ED_W2, HW_EDW2, 128, 4, 3},
      {P_VS_W1, HW_VSW1, 128, 4, 128},  {P_VS_W2, HW_VSW2, 128, 4, 1},
  };
  for (const ConvDesc& c : convs) {
    int Ntiles = (c.Nact + 15) >> 4;
    long total = (long)c.Ktiles * Ntiles * 512;
    int blocks = (int)((total + 255) / 256);
    convw_kernel<<<blocks, 256, 0, stream>>>(FIN(c.src), hws + c.dst, c.Kact,
                                             c.Ktiles, Ntiles, c.Nact);
  }

  // zero edge-aggregation accumulators (force/torque/static/count = 10N floats)
  zero_kernel<<<(10 * NN + 255) / 256, 256, 0, stream>>>(fws + OFF_FORCE, 10 * NN);

  // per-node MLPs
  NodeArgs na;
  na.nlat = FIN(IN_NLAT);
  na.w1[0] = hws + HW_NWW1; na.w2[0] = hws + HW_NWW2;
  na.w1[1] = hws + HW_IMW1; na.w2[1] = hws + HW_IMW2;
  na.w1[2] = hws + HW_IIW1; na.w2[2] = hws + HW_IIW2;
  na.w1[3] = hws + HW_EDW1; na.w2[3] = hws + HW_EDW2;
  na.w1[4] = hws + HW_VSW1; na.w2[4] = hws + HW_VSW2;
  na.b1[0] = FIN(P_NW_B1); na.b2[0] = FIN(P_NW_B2);
  na.b1[1] = FIN(P_IM_B1); na.b2[1] = FIN(P_IM_B2);
  na.b1[2] = FIN(P_II_B1); na.b2[2] = FIN(P_II_B2);
  na.b1[3] = FIN(P_ED_B1); na.b2[3] = FIN(P_ED_B2);
  na.b1[4] = FIN(P_VS_B1); na.b2[4] = FIN(P_VS_B2);
  na.outp[0] = fws + OFF_WNODE;
  na.outp[1] = fws + OFF_INVM;
  na.outp[2] = fws + OFF_INVI;
  na.outp[3] = fws + OFF_EXTDV;
  na.outp[4] = fws + OFF_VSC;
  node_kernel<<<NN / TILE_N, 256, 0, stream>>>(na);

  // per-edge pipeline
  EdgeArgs ea;
  const int* ei = (const int*)d_in[IN_EDGE_INDEX];
  ea.senders = ei;
  ea.receivers = ei + NE;
  ea.spos = FIN(IN_SPOS); ea.rpos = FIN(IN_RPOS); ea.dx = FIN(IN_DX);
  ea.eattr = FIN(IN_EATTR);
  ea.va = FIN(IN_VA); ea.vb = FIN(IN_VB); ea.vc = FIN(IN_VC);
  ea.svt = FIN(IN_SVT); ea.swt = FIN(IN_SWT);
  ea.rvt = FIN(IN_RVT); ea.rwt = FIN(IN_RWT);
  ea.nlat = FIN(IN_NLAT);
  ea.en_b1 = FIN(P_EN_B1); ea.en_b2 = FIN(P_EN_B2);
  ea.en_g = FIN(P_EN_G);   ea.en_bln = FIN(P_EN_BLN);
  ea.ee_b1 = FIN(P_EE_B1); ea.ee_b2 = FIN(P_EE_B2);
  ea.ee_g = FIN(P_EE_G);   ea.ee_bln = FIN(P_EE_BLN);
  ea.ei_b1 = FIN(P_EI_B1); ea.ei_b2 = FIN(P_EI_B2);
  ea.ei_g = FIN(P_EI_G);   ea.ei_bln = FIN(P_EI_BLN);
  ea.i1_b1 = FIN(P_I1_B1); ea.i1_b2 = FIN(P_I1_B2);
  ea.i2_b1 = FIN(P_I2_B1); ea.i2_b2 = FIN(P_I2_B2);
  ea.fs_b1 = FIN(P_FS_B1); ea.fs_b2 = FIN(P_FS_B2);
  ea.cc_b1 = FIN(P_CC_B1); ea.cc_b2 = FIN(P_CC_B2);
  ea.wENCN1 = hws + HW_ENCN1; ea.wENCN2 = hws + HW_ENCN2;
  ea.wENCE1 = hws + HW_ENCE1; ea.wENCE2 = hws + HW_ENCE2;
  ea.wENCI1 = hws + HW_ENCI1; ea.wENCI2 = hws + HW_ENCI2;
  ea.wI1a = hws + HW_I1W1; ea.wI1b = hws + HW_I1W2;
  ea.wI2a = hws + HW_I2W1; ea.wI2b = hws + HW_I2W2;
  ea.wFSa = hws + HW_FSW1; ea.wFSb = hws + HW_FSW2;
  ea.wCCa = hws + HW_CCW1; ea.wCCb = hws + HW_CCW2;
  ea.wnode = fws + OFF_WNODE;
  ea.accF = fws + OFF_FORCE;
  ea.accT = fws + OFF_TORQUE;
  ea.accS = fws + OFF_STATIC;
  ea.accC = fws + OFF_COUNT;
  ea.out_int = (float*)d_out + (size_t)9 * NN;
  edge_kernel<<<NE / TILE_E, 256, 0, stream>>>(ea);

  // finalize per-node outputs
  FinArgs fa;
  fa.force = fws + OFF_FORCE;
  fa.torque = fws + OFF_TORQUE;
  fa.stat = fws + OFF_STATIC;
  fa.count = fws + OFF_COUNT;
  fa.invm = fws + OFF_INVM;
  fa.invi = fws + OFF_INVI;
  fa.extdv = fws + OFF_EXTDV;
  fa.vsc = fws + OFF_VSC;
  fa.vel = FIN(IN_VEL);
  fa.out = (float*)d_out;
  finalize_kernel<<<(NN + 255) / 256, 256, 0, stream>>>(fa);
#undef FIN
}
